// Model_30657476559130
// MI455X (gfx1250) — matrix operand order repaired and checked by reading, not compiled, not run on hardware
//
#include <hip/hip_runtime.h>
#include <hip/hip_bf16.h>
#include <cstdint>
#include <cstddef>

// ---------------------------------------------------------------------------
// Types for CDNA5 WMMA (wave32)
// ---------------------------------------------------------------------------
typedef __bf16 bf16_t;
typedef __attribute__((ext_vector_type(16))) __bf16 v16bf;
typedef __attribute__((ext_vector_type(8)))  __bf16 v8bf;
typedef __attribute__((ext_vector_type(8)))  float  v8f;

#define TM 128
#define TN 64
#define TKK 32

// ---------------------------------------------------------------------------
// Generic GEMM:  Out[m,n] = sum_k X[m,k] * Wmat(k,n) + bias[n]  (optional ReLU)
//   wKN==0 : W stored (N,K) row-major  (conv style, W[co][ci])
//   wKN==1 : W stored (K,N) row-major  (linear style, x @ W)
// bf16 A/B with f32 accumulate via v_wmma_f32_16x16x32_bf16.
// Block = 256 threads = 8 waves; block tile 128x64; each wave owns 32x32.
// Interior B tiles staged through LDS with global_load_async_to_lds_b32.
// ---------------------------------------------------------------------------
__launch_bounds__(256)
__global__ void gemm_kernel(const float* __restrict__ X, const float* __restrict__ W,
                            const float* __restrict__ bias, float* __restrict__ Out,
                            int M, int N, int K, int wKN, int relu)
{
    __shared__ __align__(16) bf16_t As[TM][TKK + 8];   // row-major (m, k)
    __shared__ __align__(16) bf16_t Bs[TN][TKK + 8];   // col-major (n, k) -> contiguous frags
    __shared__ __align__(16) float  Bfs[TKK * TN];     // async f32 staging

    const int tid  = threadIdx.x;
    const int lane = tid & 31;
    const int wv   = tid >> 5;   // 0..7
    const int wm   = wv >> 1;    // 0..3 : 32-row strip
    const int wn   = wv & 1;     // 0..1 : 32-col strip
    const long blockM = (long)blockIdx.x * TM;
    const long blockN = (long)blockIdx.y * TN;

    const bool mFull = (blockM + TM) <= (long)M;
    const bool nFull = (blockN + TN) <= (long)N;

    const v8f zf = {0.f,0.f,0.f,0.f,0.f,0.f,0.f,0.f};
    v8f acc[2][2] = {{zf, zf}, {zf, zf}};

    const int c16 = lane & 15;
    const int hi  = lane >> 4;

    for (int k0 = 0; k0 < K; k0 += TKK) {
        const bool kFull = (k0 + TKK) <= K;

        // ---------------- stage A tile (128 x 32) ----------------------------
        if (mFull && kFull) {
#pragma unroll
            for (int t = tid; t < TM * TKK; t += 256) {
                int r = t >> 5, c = t & 31;
                As[r][c] = (bf16_t)X[(blockM + r) * (long)K + (k0 + c)];
            }
        } else {
            for (int t = tid; t < TM * TKK; t += 256) {
                int r = t >> 5, c = t & 31;
                long gm = blockM + r; int gk = k0 + c;
                float v = (gm < M && gk < K) ? X[gm * (long)K + gk] : 0.f;
                As[r][c] = (bf16_t)v;
            }
        }

        // ---------------- stage B tile (32 x 64) -----------------------------
        if (nFull && kFull) {
            // CDNA5 async path: per-lane global -> LDS copy, tracked by ASYNCcnt.
#pragma unroll
            for (int t = tid; t < TKK * TN; t += 256) {
                int r = t / TN, c = t % TN;
                const float* gp = wKN ? (W + (size_t)(k0 + r) * N + (blockN + c))
                                      : (W + (size_t)(blockN + c) * K + (k0 + r));
                unsigned ldsOff = (unsigned)(unsigned long long)(const void*)&Bfs[t];
                unsigned long long ga = (unsigned long long)gp;
                asm volatile("global_load_async_to_lds_b32 %0, %1, off"
                             :: "v"(ldsOff), "v"(ga) : "memory");
            }
            asm volatile("s_wait_asynccnt 0x0" ::: "memory");
            // each thread converts exactly the elements it staged
#pragma unroll
            for (int t = tid; t < TKK * TN; t += 256) {
                int r = t / TN, c = t % TN;
                Bs[c][r] = (bf16_t)Bfs[t];
            }
        } else {
            for (int t = tid; t < TKK * TN; t += 256) {
                int r = t / TN, c = t % TN;
                int gk = k0 + r; long gn = blockN + c;
                float v = 0.f;
                if (gk < K && gn < N)
                    v = wKN ? W[(size_t)gk * N + gn] : W[(size_t)gn * K + gk];
                Bs[c][r] = (bf16_t)v;
            }
        }

        // prefetch next K-slab of X (global_prefetch_b8)
        if (mFull && (k0 + TKK) < K) {
            long pr = blockM + (tid >> 1);
            __builtin_prefetch(&X[pr * (long)K + (k0 + TKK) + (tid & 1) * 16], 0, 0);
        }
        __syncthreads();

        // ---------------- fragments (vectorized ds_load_b128) ----------------
        v16bf afr[2], bfr[2];
#pragma unroll
        for (int i = 0; i < 2; ++i) {
            int row = wm * 32 + i * 16 + c16;
            int ko  = hi ? 16 : 0; // lanes 0-15: K 0..15 ; 16-31: K 16..31
            v8bf a0 = *(const v8bf*)&As[row][ko];
            v8bf a1 = *(const v8bf*)&As[row][ko + 8];
            afr[i] = __builtin_shufflevector(a0, a1, 0,1,2,3,4,5,6,7,8,9,10,11,12,13,14,15);
        }
#pragma unroll
        for (int j = 0; j < 2; ++j) {
            int col = wn * 32 + j * 16 + c16;
            int kb  = hi ? 16 : 0; // lanes 0-15: K 0..15 ; 16-31: K 16..31
            v8bf b0 = *(const v8bf*)&Bs[col][kb];
            v8bf b1 = *(const v8bf*)&Bs[col][kb + 8];
            bfr[j] = __builtin_shufflevector(b0, b1, 0,1,2,3,4,5,6,7,8,9,10,11,12,13,14,15);
        }
#pragma unroll
        for (int i = 0; i < 2; ++i)
#pragma unroll
            for (int j = 0; j < 2; ++j)
                acc[i][j] = __builtin_amdgcn_wmma_f32_16x16x32_bf16(
                    false, afr[i], false, bfr[j], (short)0, acc[i][j], false, false);
        __syncthreads();
    }

    // Store: C/D layout — VGPR r holds row r (lanes 0-15) / row r+8 (lanes 16-31)
    const int rOff = hi ? 8 : 0;
#pragma unroll
    for (int i = 0; i < 2; ++i) {
#pragma unroll
        for (int j = 0; j < 2; ++j) {
            long gcol = blockN + wn * 32 + j * 16 + c16;
            if (gcol >= N) continue;
            float bv = bias ? bias[gcol] : 0.f;
#pragma unroll
            for (int r = 0; r < 8; ++r) {
                long grow = blockM + wm * 32 + i * 16 + rOff + r;
                if (grow < M) {
                    float v = acc[i][j][r] + bv;
                    if (relu && v < 0.f) v = 0.f;
                    Out[grow * (long)N + gcol] = v;
                }
            }
        }
    }
}

// ---------------------------------------------------------------------------
// Per-column mean / rstd over M rows  (training-mode BN statistics)
// ---------------------------------------------------------------------------
__global__ void colstats_kernel(const float* __restrict__ X, int M, int N,
                                float* __restrict__ mean, float* __restrict__ rstd)
{
    const int n   = blockIdx.x;
    const int tid = threadIdx.x;
    float s = 0.f, s2 = 0.f;
    for (int m = tid; m < M; m += 256) {
        float v = X[(size_t)m * N + n];
        s += v; s2 += v * v;
    }
    __shared__ float ss[256], sq[256];
    ss[tid] = s; sq[tid] = s2;
    __syncthreads();
    for (int st = 128; st > 0; st >>= 1) {
        if (tid < st) { ss[tid] += ss[tid + st]; sq[tid] += sq[tid + st]; }
        __syncthreads();
    }
    if (tid == 0) {
        float mu  = ss[0] / (float)M;
        float var = sq[0] / (float)M - mu * mu;
        mean[n] = mu;
        rstd[n] = rsqrtf(var + 1e-5f);
    }
}

__global__ void bnrelu_kernel(float* __restrict__ X, const float* __restrict__ mean,
                              const float* __restrict__ rstd, const float* __restrict__ g,
                              const float* __restrict__ b, size_t total, int N, int relu)
{
    size_t i = (size_t)blockIdx.x * blockDim.x + threadIdx.x;
    if (i >= total) return;
    int n = (int)(i % (size_t)N);
    float v = (X[i] - mean[n]) * rstd[n] * g[n] + b[n];
    if (relu && v < 0.f) v = 0.f;
    X[i] = v;
}

// ---------------------------------------------------------------------------
// Farthest-point sampling: one block per batch, LDS-resident distance array
// ---------------------------------------------------------------------------
__launch_bounds__(256)
__global__ void fps_kernel(const float* __restrict__ xyz, int Npts, int npoint,
                           int* __restrict__ idx)
{
    __shared__ float dists[8192];
    __shared__ float bestv[256];
    __shared__ int   besti[256];
    __shared__ float last[3];
    __shared__ int   lastIdx;

    const int b   = blockIdx.x;
    const int tid = threadIdx.x;
    const float* p = xyz + (size_t)b * Npts * 3;

    for (int i = tid; i < Npts; i += 256) dists[i] = 1e10f;
    if (tid == 0) { lastIdx = 0; idx[b * npoint] = 0; }
    __syncthreads();

    for (int it = 1; it < npoint; ++it) {
        if (tid == 0) {
            int li = lastIdx;
            last[0] = p[li * 3]; last[1] = p[li * 3 + 1]; last[2] = p[li * 3 + 2];
        }
        __syncthreads();
        float lx = last[0], ly = last[1], lz = last[2];
        float bm = -1.f; int bi = 0;
        for (int i = tid; i < Npts; i += 256) {
            float dx = p[i * 3] - lx, dy = p[i * 3 + 1] - ly, dz = p[i * 3 + 2] - lz;
            float d = dx * dx + dy * dy + dz * dz;
            float cur = dists[i];
            d = (d < cur) ? d : cur;
            dists[i] = d;
            if (d > bm) { bm = d; bi = i; }      // keeps lowest index on ties
        }
        bestv[tid] = bm; besti[tid] = bi;
        __syncthreads();
        for (int st = 128; st > 0; st >>= 1) {
            if (tid < st) {
                if (bestv[tid + st] > bestv[tid] ||
                    (bestv[tid + st] == bestv[tid] && besti[tid + st] < besti[tid])) {
                    bestv[tid] = bestv[tid + st];
                    besti[tid] = besti[tid + st];
                }
            }
            __syncthreads();
        }
        if (tid == 0) { lastIdx = besti[0]; idx[b * npoint + it] = besti[0]; }
        __syncthreads();
    }
}

// ---------------------------------------------------------------------------
// Gather xyz rows by per-batch index list: out[(b*S+s),:] = xyz[b, idx[b*S+s], :]
// ---------------------------------------------------------------------------
__global__ void gather_xyz_kernel(const float* __restrict__ xyz, const int* __restrict__ idx,
                                  float* __restrict__ out, int Npts, int S, int Btot)
{
    int t = blockIdx.x * blockDim.x + threadIdx.x;
    if (t >= Btot * S) return;
    int b = t / S;
    int j = idx[t];
    const float* src = xyz + ((size_t)b * Npts + j) * 3;
    out[t * 3 + 0] = src[0];
    out[t * 3 + 1] = src[1];
    out[t * 3 + 2] = src[2];
}

// ---------------------------------------------------------------------------
// Ball query: first-K indices with d2<=r2 in ascending order (== ref's sort trick)
// ---------------------------------------------------------------------------
__global__ void ball_query_kernel(const float* __restrict__ xyz, const float* __restrict__ nxyz,
                                  int* __restrict__ gidx, int Btot, int Npts, int S, int Kn,
                                  float r2)
{
    int t = blockIdx.x * blockDim.x + threadIdx.x;
    if (t >= Btot * S) return;
    int b = t / S;
    float qx = nxyz[t * 3], qy = nxyz[t * 3 + 1], qz = nxyz[t * 3 + 2];
    const float* p = xyz + (size_t)b * Npts * 3;
    int cnt = 0, first = 0;
    for (int j = 0; j < Npts && cnt < Kn; ++j) {
        float dx = p[j * 3] - qx, dy = p[j * 3 + 1] - qy, dz = p[j * 3 + 2] - qz;
        float d = dx * dx + dy * dy + dz * dz;
        if (d <= r2) {
            if (cnt == 0) first = j;
            gidx[(size_t)t * Kn + cnt] = j;
            ++cnt;
        }
    }
    int fill = (cnt > 0) ? first : 0;
    for (int k = cnt; k < Kn; ++k) gidx[(size_t)t * Kn + k] = fill;
}

// ---------------------------------------------------------------------------
// Build grouped rows: out[(b,s,k), 0:3] = xyz[idx]-new_xyz ; out[..,3:3+C]=feats[idx]
// feats stored rows = (B*Npts) x C
// ---------------------------------------------------------------------------
__global__ void group_gather_kernel(const float* __restrict__ xyz, const float* __restrict__ nxyz,
                                    const float* __restrict__ feats, const int* __restrict__ gidx,
                                    float* __restrict__ out, int Npts, int S, int Kn, int C,
                                    size_t total)
{
    size_t i = (size_t)blockIdx.x * blockDim.x + threadIdx.x;
    if (i >= total) return;
    int CT = 3 + C;
    size_t row = i / CT;
    int c = (int)(i % (size_t)CT);
    size_t bs = row / Kn;
    int b = (int)(bs / S);
    int j = gidx[row];
    float v;
    if (c < 3) v = xyz[((size_t)b * Npts + j) * 3 + c] - nxyz[bs * 3 + c];
    else       v = feats[((size_t)b * Npts + j) * C + (c - 3)];
    out[i] = v;
}

// ---------------------------------------------------------------------------
// Max-pool over sample dim: in (rowsOut*Kn, C) -> out (rowsOut, C)
// ---------------------------------------------------------------------------
__global__ void maxpool_kernel(const float* __restrict__ in, float* __restrict__ out,
                               int rowsOut, int Kn, int C)
{
    size_t i = (size_t)blockIdx.x * blockDim.x + threadIdx.x;
    if (i >= (size_t)rowsOut * C) return;
    size_t ro = i / C;
    int c = (int)(i % (size_t)C);
    float m = -3.4e38f;
    for (int k = 0; k < Kn; ++k) {
        float v = in[((size_t)ro * Kn + k) * C + c];
        if (v > m) m = v;
    }
    out[i] = m;
}

// ---------------------------------------------------------------------------
// Column-range copy (for concat / slice)
// ---------------------------------------------------------------------------
__global__ void copy_cols_kernel(const float* __restrict__ src, int srcLD, int srcOff,
                                 float* __restrict__ dst, int dstLD, int dstOff,
                                 int Mrows, int Cc)
{
    size_t i = (size_t)blockIdx.x * blockDim.x + threadIdx.x;
    if (i >= (size_t)Mrows * Cc) return;
    size_t m = i / Cc;
    int c = (int)(i % (size_t)Cc);
    dst[m * dstLD + dstOff + c] = src[m * srcLD + srcOff + c];
}

__global__ void copy_kernel(const float* __restrict__ s, float* __restrict__ d, int n)
{
    int i = blockIdx.x * blockDim.x + threadIdx.x;
    if (i < n) d[i] = s[i];
}

// ---------------------------------------------------------------------------
// rebuild rows (B*NC, 1027): [glob(b) | coarse(b,n,:)]
// ---------------------------------------------------------------------------
__global__ void rebuild_kernel(const float* __restrict__ glob, const float* __restrict__ coarse,
                               float* __restrict__ out, size_t total)
{
    size_t i = (size_t)blockIdx.x * blockDim.x + threadIdx.x;
    if (i >= total) return;
    int c = (int)(i % 1027);
    size_t row = i / 1027;
    int b = (int)(row / 384);
    out[i] = (c < 1024) ? glob[(size_t)b * 1024 + c] : coarse[row * 3 + (c - 1024)];
}

// ---------------------------------------------------------------------------
// fold chain input:  rows = (b*NC+n)*16 + l
//   useSeed: [tile(lin)[l], repeat(lin)[l], rf(bn)]  (CT = 2+1024)
//   else   : [fd1(row,0:3), rf(bn)]                  (CT = 3+1024)
// ---------------------------------------------------------------------------
__global__ void fold_cat_kernel(const float* __restrict__ rf, const float* __restrict__ fd1,
                                float* __restrict__ out, int useSeed, size_t total, int CT)
{
    size_t i = (size_t)blockIdx.x * blockDim.x + threadIdx.x;
    if (i >= total) return;
    int c = (int)(i % (size_t)CT);
    size_t row = i / CT;
    int l = (int)(row & 15);
    size_t bn = row >> 4;
    int Cx = useSeed ? 2 : 3;
    float v;
    if (c < Cx) {
        if (useSeed) {
            const float lv0 = -1.f, lv1 = -1.f / 3.f, lv2 = 1.f / 3.f, lv3 = 1.f;
            int sel = (c == 0) ? (l & 3) : (l >> 2);
            v = (sel == 0) ? lv0 : (sel == 1) ? lv1 : (sel == 2) ? lv2 : lv3;
        } else {
            v = fd1[row * 3 + c];
        }
    } else {
        v = rf[bn * 1024 + (c - Cx)];
    }
    out[i] = v;
}

// ---------------------------------------------------------------------------
// pc_c = pc_input - center ; also fills first 8192 rows of pc_out region
// ---------------------------------------------------------------------------
__global__ void pcc_kernel(const float* __restrict__ pc, const float* __restrict__ center,
                           float* __restrict__ pc_c, float* __restrict__ pcOut)
{
    size_t i = (size_t)blockIdx.x * blockDim.x + threadIdx.x;
    if (i >= (size_t)8 * 8192 * 3) return;
    int d = (int)(i % 3);
    size_t r = i / 3;
    int b = (int)(r / 8192);
    int n = (int)(r % 8192);
    float v = pc[i] - center[b * 3 + d];
    pc_c[i] = v;
    pcOut[((size_t)b * 14336 + n) * 3 + d] = v;
}

// ---------------------------------------------------------------------------
// coarse_out (B,512,3) = [coarse(B,384,3) | gather(pc_c, sparse_idx)(B,128,3)]
// ---------------------------------------------------------------------------
__global__ void coarseout_kernel(const float* __restrict__ coarse, const float* __restrict__ pc_c,
                                 const int* __restrict__ sidx, float* __restrict__ co)
{
    size_t i = (size_t)blockIdx.x * blockDim.x + threadIdx.x;
    if (i >= (size_t)8 * 512 * 3) return;
    int d = (int)(i % 3);
    size_t r = i / 3;
    int b = (int)(r / 512);
    int n = (int)(r % 512);
    if (n < 384) co[i] = coarse[((size_t)b * 384 + n) * 3 + d];
    else {
        int j = sidx[b * 128 + (n - 384)];
        co[i] = pc_c[((size_t)b * 8192 + j) * 3 + d];
    }
}

// ---------------------------------------------------------------------------
// fine = rel + coarse broadcast; write fine region + tail of pc_out
// ---------------------------------------------------------------------------
__global__ void fine_kernel(const float* __restrict__ rel, const float* __restrict__ coarse,
                            float* __restrict__ fineOut, float* __restrict__ pcOut)
{
    size_t i = (size_t)blockIdx.x * blockDim.x + threadIdx.x;
    if (i >= (size_t)49152 * 3) return;
    int d = (int)(i % 3);
    size_t row = i / 3;
    int l = (int)(row & 15);
    size_t bn = row >> 4;
    int b = (int)(bn / 384);
    int n = (int)(bn % 384);
    float v = rel[i] + coarse[bn * 3 + d];
    size_t pi = (size_t)n * 16 + l;
    fineOut[((size_t)b * 6144 + pi) * 3 + d] = v;
    pcOut[((size_t)b * 14336 + 8192 + pi) * 3 + d] = v;
}

// ---------------------------------------------------------------------------
// Host-side helpers
// ---------------------------------------------------------------------------
static void launch_gemm(const float* X, const float* W, const float* b, float* Y,
                        int M, int N, int K, int wKN, int relu, hipStream_t s)
{
    dim3 g((unsigned)((M + TM - 1) / TM), (unsigned)((N + TN - 1) / TN));
    gemm_kernel<<<g, 256, 0, s>>>(X, W, b, Y, M, N, K, wKN, relu);
}

static void launch_bn(float* X, int M, int N, const float* g, const float* b,
                      float* mean, float* rstd, int relu, hipStream_t s)
{
    colstats_kernel<<<N, 256, 0, s>>>(X, M, N, mean, rstd);
    size_t tot = (size_t)M * N;
    bnrelu_kernel<<<(unsigned)((tot + 255) / 256), 256, 0, s>>>(X, mean, rstd, g, b, tot, N, relu);
}

struct Arena { char* p; size_t off; };
static float* af(Arena& a, size_t n) {
    float* r = (float*)(a.p + a.off);
    a.off += ((n * sizeof(float) + 255) / 256) * 256;
    return r;
}
static int* ai(Arena& a, size_t n) {
    int* r = (int*)(a.p + a.off);
    a.off += ((n * sizeof(int) + 255) / 256) * 256;
    return r;
}

// ---------------------------------------------------------------------------
// Orchestration. Params flattened in jax pytree order (dict keys sorted):
// center_map, conv1, conv2, conv3, fold1, fold2, mlp, offset_map, reduce_map,
// sa1, sa2, sa3, sa4, sa_sep.
// ---------------------------------------------------------------------------
extern "C" void kernel_launch(void* const* d_in, const int* in_sizes, int n_in,
                              void* d_out, int out_size, void* d_ws, size_t ws_size,
                              hipStream_t stream)
{
    (void)in_sizes; (void)n_in; (void)out_size; (void)ws_size;
    auto F = [&](int i) { return (const float*)d_in[i]; };

    const float* pc_input = F(0);
    // center_map
    const float *cmW1=F(1),*cmB1=F(2),*cmG1=F(3),*cmT1=F(4);
    const float *cmW2=F(5),*cmB2=F(6),*cmG2=F(7),*cmT2=F(8);
    const float *cmW3=F(9),*cmB3=F(10);
    // conv1..3
    const float *c1W=F(11),*c1B=F(12),*c1G=F(13),*c1T=F(14);
    const float *c2W=F(15),*c2B=F(16),*c2G=F(17),*c2T=F(18);
    const float *c3W=F(19),*c3B=F(20);
    // fold1
    const float *f1W1=F(21),*f1B1=F(22),*f1G1=F(23),*f1T1=F(24);
    const float *f1W2=F(25),*f1B2=F(26),*f1G2=F(27),*f1T2=F(28);
    const float *f1W3=F(29),*f1B3=F(30);
    // fold2
    const float *f2W1=F(31),*f2B1=F(32),*f2G1=F(33),*f2T1=F(34);
    const float *f2W2=F(35),*f2B2=F(36),*f2G2=F(37),*f2T2=F(38);
    const float *f2W3=F(39),*f2B3=F(40);
    // mlp
    const float *mW1=F(41),*mB1=F(42),*mW2=F(43),*mB2=F(44),*mW3=F(45),*mB3=F(46);
    // offset_map / reduce_map
    const float *offW=F(47),*offB=F(48),*redW=F(49),*redB=F(50);
    // sa1..sa_sep
    const float *s1W1=F(51),*s1B1=F(52),*s1G1=F(53),*s1T1=F(54);
    const float *s1W2=F(55),*s1B2=F(56),*s1G2=F(57),*s1T2=F(58);
    const float *s1W3=F(59),*s1B3=F(60),*s1G3=F(61),*s1T3=F(62);
    const float *s2W1=F(63),*s2B1=F(64),*s2G1=F(65),*s2T1=F(66);
    const float *s2W2=F(67),*s2B2=F(68),*s2G2=F(69),*s2T2=F(70);
    const float *s2W3=F(71),*s2B3=F(72),*s2G3=F(73),*s2T3=F(74);
    const float *s3W1=F(75),*s3B1=F(76),*s3G1=F(77),*s3T1=F(78);
    const float *s3W2=F(79),*s3B2=F(80),*s3G2=F(81),*s3T2=F(82);
    const float *s3W3=F(83),*s3B3=F(84),*s3G3=F(85),*s3T3=F(86);
    const float *s4W1=F(87),*s4B1=F(88),*s4G1=F(89),*s4T1=F(90);
    const float *s4W2=F(91),*s4B2=F(92),*s4G2=F(93),*s4T2=F(94);
    const float *s4W3=F(95),*s4B3=F(96),*s4G3=F(97),*s4T3=F(98);
    const float *spW1=F(99),*spB1=F(100),*spG1=F(101),*spT1=F(102);
    const float *spW2=F(103),*spB2=F(104),*spG2=F(105),*spT2=F(106);
    const float *spW3=F(107),*spB3=F(108),*spG3=F(109),*spT3=F(110);

    const int Bn = 8, Np = 8192, NC = 384;
    float* outp = (float*)d_out;
    float* pcOut     = outp + 24;                 // (8,14336,3)
    float* coarseOut = pcOut + (size_t)8*14336*3; // (8,512,3)
    float* fineOut   = coarseOut + (size_t)8*512*3;

    Arena A{ (char*)d_ws, 0 };
    float* bufA   = af(A, 50500000);   // up to 49152 x 1027
    float* bufB   = af(A, 26000000);   // up to 49152 x 512
    float* pc_c   = af(A, (size_t)Bn*Np*3);
    float* xyz1   = af(A, (size_t)Bn*512*3);
    float* feats1 = af(A, (size_t)Bn*512*64);
    float* xyz2   = af(A, (size_t)Bn*256*3);
    float* feats2 = af(A, (size_t)Bn*256*256);
    float* xyz3   = af(A, (size_t)Bn*128*3);
    float* feats3 = af(A, (size_t)Bn*128*512);
    float* feats4 = af(A, (size_t)Bn*1024);
    float* sepxyz = af(A, (size_t)Bn*4*3);
    float* sepf   = af(A, (size_t)Bn*1024);
    float* fvec   = af(A, (size_t)Bn*2048);
    float* out1   = af(A, (size_t)Bn*1024);
    float* cat2   = af(A, (size_t)Bn*2048);
    float* out2   = af(A, (size_t)Bn*2048);
    float* out3   = af(A, (size_t)Bn*2048);
    float* cmin   = af(A, (size_t)Bn*1024);
    float* globb  = af(A, (size_t)Bn*1024);
    float* t512   = af(A, (size_t)Bn*512);
    float* t256   = af(A, (size_t)Bn*256);
    float* centerB= af(A, (size_t)Bn*3);
    float* globc  = af(A, (size_t)Bn*1027);
    float* glob2  = af(A, (size_t)Bn*1024);
    float* h1     = af(A, (size_t)Bn*1024);
    float* h2     = af(A, (size_t)Bn*1024);
    float* coarse = af(A, (size_t)Bn*NC*3);
    float* rebuild= af(A, (size_t)Bn*NC*1027);
    float* rf     = af(A, (size_t)Bn*NC*1024);
    float* fd1    = af(A, (size_t)Bn*NC*16*3);
    float* rel    = af(A, (size_t)Bn*NC*16*3);
    float* meanB  = af(A, 2048);
    float* rstdB  = af(A, 2048);
    int* fpsIdx   = ai(A, (size_t)Bn*512);
    int* gidx     = ai(A, (size_t)Bn*512*32);
    int* sepIdx   = ai(A, (size_t)Bn*4);
    int* spIdx    = ai(A, (size_t)Bn*128);

    auto GS = [](size_t tot) { return (unsigned)((tot + 255) / 256); };

    // ---------------- SA1: 8192 -> 512 pts, r=0.02, K=32, ch 3->32->32->64 ----
    fps_kernel<<<Bn, 256, 0, stream>>>(pc_input, Np, 512, fpsIdx);
    gather_xyz_kernel<<<GS((size_t)Bn*512), 256, 0, stream>>>(pc_input, fpsIdx, xyz1, Np, 512, Bn);
    ball_query_kernel<<<GS((size_t)Bn*512), 256, 0, stream>>>(pc_input, xyz1, gidx, Bn, Np, 512, 32, 0.02f*0.02f);
    {
        size_t rows = (size_t)Bn*512*32; size_t tot = rows*3;
        group_gather_kernel<<<GS(tot), 256, 0, stream>>>(pc_input, xyz1, nullptr, gidx, bufA, Np, 512, 32, 0, tot);
        int M = (int)rows;
        launch_gemm(bufA, s1W1, s1B1, bufB, M, 32, 3, 0, 0, stream);  launch_bn(bufB, M, 32, s1G1, s1T1, meanB, rstdB, 1, stream);
        launch_gemm(bufB, s1W2, s1B2, bufA, M, 32, 32, 0, 0, stream); launch_bn(bufA, M, 32, s1G2, s1T2, meanB, rstdB, 1, stream);
        launch_gemm(bufA, s1W3, s1B3, bufB, M, 64, 32, 0, 0, stream); launch_bn(bufB, M, 64, s1G3, s1T3, meanB, rstdB, 1, stream);
        maxpool_kernel<<<GS((size_t)Bn*512*64), 256, 0, stream>>>(bufB, feats1, Bn*512, 32, 64);
    }

    // ---------------- SA2: 512 -> 256, r=0.04, K=16, ch 67->128->128->256 -----
    fps_kernel<<<Bn, 256, 0, stream>>>(xyz1, 512, 256, fpsIdx);
    gather_xyz_kernel<<<GS((size_t)Bn*256), 256, 0, stream>>>(xyz1, fpsIdx, xyz2, 512, 256, Bn);
    ball_query_kernel<<<GS((size_t)Bn*256), 256, 0, stream>>>(xyz1, xyz2, gidx, Bn, 512, 256, 16, 0.04f*0.04f);
    {
        size_t rows = (size_t)Bn*256*16; size_t tot = rows*67;
        group_gather_kernel<<<GS(tot), 256, 0, stream>>>(xyz1, xyz2, feats1, gidx, bufA, 512, 256, 16, 64, tot);
        int M = (int)rows;
        launch_gemm(bufA, s2W1, s2B1, bufB, M, 128, 67, 0, 0, stream);  launch_bn(bufB, M, 128, s2G1, s2T1, meanB, rstdB, 1, stream);
        launch_gemm(bufB, s2W2, s2B2, bufA, M, 128, 128, 0, 0, stream); launch_bn(bufA, M, 128, s2G2, s2T2, meanB, rstdB, 1, stream);
        launch_gemm(bufA, s2W3, s2B3, bufB, M, 256, 128, 0, 0, stream); launch_bn(bufB, M, 256, s2G3, s2T3, meanB, rstdB, 1, stream);
        maxpool_kernel<<<GS((size_t)Bn*256*256), 256, 0, stream>>>(bufB, feats2, Bn*256, 16, 256);
    }

    // ---------------- SA3: 256 -> 128, r=0.08, K=16, ch 259->256->512->512 ----
    fps_kernel<<<Bn, 256, 0, stream>>>(xyz2, 256, 128, fpsIdx);
    gather_xyz_kernel<<<GS((size_t)Bn*128), 256, 0, stream>>>(xyz2, fpsIdx, xyz3, 256, 128, Bn);
    ball_query_kernel<<<GS((size_t)Bn*128), 256, 0, stream>>>(xyz2, xyz3, gidx, Bn, 256, 128, 16, 0.08f*0.08f);
    {
        size_t rows = (size_t)Bn*128*16; size_t tot = rows*259;
        group_gather_kernel<<<GS(tot), 256, 0, stream>>>(xyz2, xyz3, feats2, gidx, bufA, 256, 128, 16, 256, tot);
        int M = (int)rows;
        launch_gemm(bufA, s3W1, s3B1, bufB, M, 256, 259, 0, 0, stream); launch_bn(bufB, M, 256, s3G1, s3T1, meanB, rstdB, 1, stream);
        launch_gemm(bufB, s3W2, s3B2, bufA, M, 512, 256, 0, 0, stream); launch_bn(bufA, M, 512, s3G2, s3T2, meanB, rstdB, 1, stream);
        launch_gemm(bufA, s3W3, s3B3, bufB, M, 512, 512, 0, 0, stream); launch_bn(bufB, M, 512, s3G3, s3T3, meanB, rstdB, 1, stream);
        maxpool_kernel<<<GS((size_t)Bn*128*512), 256, 0, stream>>>(bufB, feats3, Bn*128, 16, 512);
    }

    // ---------------- SA4: group_all (128 pts), ch 515->512->1024->1024 -------
    {
        int M = Bn*128;
        copy_cols_kernel<<<GS((size_t)M*3), 256, 0, stream>>>(xyz3, 3, 0, bufA, 515, 0, M, 3);
        copy_cols_kernel<<<GS((size_t)M*512), 256, 0, stream>>>(feats3, 512, 0, bufA, 515, 3, M, 512);
        launch_gemm(bufA, s4W1, s4B1, bufB, M, 512, 515, 0, 0, stream);  launch_bn(bufB, M, 512, s4G1, s4T1, meanB, rstdB, 1, stream);
        launch_gemm(bufB, s4W2, s4B2, bufA, M, 1024, 512, 0, 0, stream); launch_bn(bufA, M, 1024, s4G2, s4T2, meanB, rstdB, 1, stream);
        launch_gemm(bufA, s4W3, s4B3, bufB, M, 1024, 1024, 0, 0, stream);launch_bn(bufB, M, 1024, s4G3, s4T3, meanB, rstdB, 1, stream);
        maxpool_kernel<<<GS((size_t)Bn*1024), 256, 0, stream>>>(bufB, feats4, Bn, 128, 1024);
    }

    // ---------------- SA_sep: FPS(pc,4) -> group_all, ch 3->256->512->1024 ----
    fps_kernel<<<Bn, 256, 0, stream>>>(pc_input, Np, 4, sepIdx);
    gather_xyz_kernel<<<GS((size_t)Bn*4), 256, 0, stream>>>(pc_input, sepIdx, sepxyz, Np, 4, Bn);
    {
        int M = Bn*4;
        launch_gemm(sepxyz, spW1, spB1, bufA, M, 256, 3, 0, 0, stream);  launch_bn(bufA, M, 256, spG1, spT1, meanB, rstdB, 1, stream);
        launch_gemm(bufA, spW2, spB2, bufB, M, 512, 256, 0, 0, stream);  launch_bn(bufB, M, 512, spG2, spT2, meanB, rstdB, 1, stream);
        launch_gemm(bufB, spW3, spB3, bufA, M, 1024, 512, 0, 0, stream); launch_bn(bufA, M, 1024, spG3, spT3, meanB, rstdB, 1, stream);
        maxpool_kernel<<<GS((size_t)Bn*1024), 256, 0, stream>>>(bufA, sepf, Bn, 4, 1024);
    }

    // ---------------- conv stack ---------------------------------------------
    copy_cols_kernel<<<GS((size_t)Bn*1024), 256, 0, stream>>>(feats4, 1024, 0, fvec, 2048, 0,    Bn, 1024);
    copy_cols_kernel<<<GS((size_t)Bn*1024), 256, 0, stream>>>(sepf,   1024, 0, fvec, 2048, 1024, Bn, 1024);
    launch_gemm(fvec, c1W, c1B, out1, Bn, 1024, 2048, 0, 0, stream); launch_bn(out1, Bn, 1024, c1G, c1T, meanB, rstdB, 1, stream);
    copy_cols_kernel<<<GS((size_t)Bn*1024), 256, 0, stream>>>(out1, 1024, 0, cat2, 2048, 0,    Bn, 1024);
    copy_cols_kernel<<<GS((size_t)Bn*1024), 256, 0, stream>>>(sepf, 1024, 0, cat2, 2048, 1024, Bn, 1024);
    launch_gemm(cat2, c2W, c2B, out2, Bn, 2048, 2048, 0, 0, stream); launch_bn(out2, Bn, 2048, c2G, c2T, meanB, rstdB, 1, stream);
    launch_gemm(out2, c3W, c3B, out3, Bn, 2048, 2048, 0, 0, stream);

    // ---------------- heads --------------------------------------------------
    copy_cols_kernel<<<GS((size_t)Bn*1024), 256, 0, stream>>>(out3, 2048, 0,    cmin,  1024, 0, Bn, 1024);
    copy_cols_kernel<<<GS((size_t)Bn*1024), 256, 0, stream>>>(out3, 2048, 1024, globb, 1024, 0, Bn, 1024);
    launch_gemm(cmin, cmW1, cmB1, t512, Bn, 512, 1024, 1, 0, stream); launch_bn(t512, Bn, 512, cmG1, cmT1, meanB, rstdB, 1, stream);
    launch_gemm(t512, cmW2, cmB2, t256, Bn, 256, 512, 1, 0, stream);  launch_bn(t256, Bn, 256, cmG2, cmT2, meanB, rstdB, 1, stream);
    launch_gemm(t256, cmW3, cmB3, centerB, Bn, 3, 256, 1, 0, stream);

    copy_cols_kernel<<<GS((size_t)Bn*1024), 256, 0, stream>>>(globb,   1024, 0, globc, 1027, 0,    Bn, 1024);
    copy_cols_kernel<<<GS((size_t)Bn*3),    256, 0, stream>>>(centerB, 3,    0, globc, 1027, 1024, Bn, 3);
    launch_gemm(globc, offW, offB, glob2, Bn, 1024, 1027, 1, 0, stream);

    launch_gemm(glob2, mW1, mB1, h1, Bn, 1024, 1024, 1, 1, stream);
    launch_gemm(h1,    mW2, mB2, h2, Bn, 1024, 1024, 1, 1, stream);
    launch_gemm(h2,    mW3, mB3, coarse, Bn, 1152, 1024, 1, 0, stream);

    // ---------------- rebuild + reduce_map -----------------------------------
    {
        size_t tot = (size_t)Bn*NC*1027;
        rebuild_kernel<<<GS(tot), 256, 0, stream>>>(glob2, coarse, rebuild, tot);
        launch_gemm(rebuild, redW, redB, rf, Bn*NC, 1024, 1027, 1, 0, stream);
    }

    // ---------------- folding ------------------------------------------------
    {
        int Mf = Bn*NC*16; // 49152
        size_t tot1 = (size_t)Mf*1026;
        fold_cat_kernel<<<GS(tot1), 256, 0, stream>>>(rf, nullptr, bufA, 1, tot1, 1026);
        launch_gemm(bufA, f1W1, f1B1, bufB, Mf, 512, 1026, 0, 0, stream); launch_bn(bufB, Mf, 512, f1G1, f1T1, meanB, rstdB, 1, stream);
        launch_gemm(bufB, f1W2, f1B2, bufA, Mf, 256, 512, 0, 0, stream);  launch_bn(bufA, Mf, 256, f1G2, f1T2, meanB, rstdB, 1, stream);
        launch_gemm(bufA, f1W3, f1B3, fd1,  Mf, 3, 256, 0, 0, stream);

        size_t tot2 = (size_t)Mf*1027;
        fold_cat_kernel<<<GS(tot2), 256, 0, stream>>>(rf, fd1, bufA, 0, tot2, 1027);
        launch_gemm(bufA, f2W1, f2B1, bufB, Mf, 512, 1027, 0, 0, stream); launch_bn(bufB, Mf, 512, f2G1, f2T1, meanB, rstdB, 1, stream);
        launch_gemm(bufB, f2W2, f2B2, bufA, Mf, 256, 512, 0, 0, stream);  launch_bn(bufA, Mf, 256, f2G2, f2T2, meanB, rstdB, 1, stream);
        launch_gemm(bufA, f2W3, f2B3, rel,  Mf, 3, 256, 0, 0, stream);
    }

    // ---------------- outputs ------------------------------------------------
    copy_kernel<<<1, 32, 0, stream>>>(centerB, outp, 24);
    pcc_kernel<<<GS((size_t)Bn*Np*3), 256, 0, stream>>>(pc_input, centerB, pc_c, pcOut);
    fps_kernel<<<Bn, 256, 0, stream>>>(pc_c, Np, 128, spIdx);
    coarseout_kernel<<<GS((size_t)Bn*512*3), 256, 0, stream>>>(coarse, pc_c, spIdx, coarseOut);
    fine_kernel<<<GS((size_t)49152*3), 256, 0, stream>>>(rel, coarse, fineOut, pcOut);
}
